// Polynormer_20349555048608
// MI455X (gfx1250) — matrix operand order repaired and checked by reading, not compiled, not run on hardware
//
#include <hip/hip_runtime.h>

#define DD 256
#define HH 4
#define CC 64

typedef unsigned short u16;
typedef __attribute__((ext_vector_type(16))) __bf16 v16bf;
typedef __attribute__((ext_vector_type(8)))  float v8f;
typedef __attribute__((ext_vector_type(4)))  unsigned int u32x4;
typedef __attribute__((ext_vector_type(4)))  unsigned short u16x4;

__device__ __forceinline__ u16 f2bf(float f) {
    unsigned u = __float_as_uint(f);
    u += 0x7FFFu + ((u >> 16) & 1u);   // round-to-nearest-even
    return (u16)(u >> 16);
}
// monotonic float->uint key for atomicMax-based segment max
__device__ __forceinline__ unsigned fkey(float f) {
    unsigned u = __float_as_uint(f);
    return (u & 0x80000000u) ? ~u : (u | 0x80000000u);
}
__device__ __forceinline__ float fkey_inv(unsigned k) {
    return __uint_as_float((k & 0x80000000u) ? (k & 0x7FFFFFFFu) : ~k);
}

__global__ void cvt_bf16_kernel(const float* __restrict__ s, u16* __restrict__ d, size_t n) {
    size_t i = (size_t)blockIdx.x * blockDim.x + threadIdx.x;
    if (i < n) d[i] = f2bf(s[i]);
}
__global__ void fill_f32_kernel(float* __restrict__ p, float v, size_t n) {
    size_t i = (size_t)blockIdx.x * blockDim.x + threadIdx.x;
    if (i < n) p[i] = v;
}
__global__ void fill_u32_kernel(unsigned* __restrict__ p, unsigned v, size_t n) {
    size_t i = (size_t)blockIdx.x * blockDim.x + threadIdx.x;
    if (i < n) p[i] = v;
}

// C[M,Nout] = act( (addC ? C : 0) + A(bf16)[M,K] @ W(bf16)[K,Nout] + bias0 + bias1 )
// Optional bf16 copy to Cb. Block: 8 waves x 16 rows = 128 rows, 64-column LDS
// weight panel. Each wave register-blocks a 16x64 tile: one A fragment per
// k-step reused against 4 B fragments -> 4 v_wmma_f32_16x16x32_bf16 per step.
__global__ __launch_bounds__(256)
void gemm_bf16_wmma(const u16* __restrict__ A, const u16* __restrict__ W,
                    float* __restrict__ C, u16* __restrict__ Cb,
                    const float* __restrict__ bias0, const float* __restrict__ bias1,
                    int M, int K, int Nout, int addC, int doRelu)
{
    __shared__ u16 WT[64][264];             // [n][k], +8 pad -> conflict-free b128 reads
    const int n0 = blockIdx.y * 64;
    for (int idx = threadIdx.x; idx < K * 16; idx += 256) {
        int k = idx >> 4, nq = (idx & 15) << 2;            // 4 cols per thread, 8B reads
        u16x4 w4 = *(const u16x4*)(W + (size_t)k * Nout + n0 + nq);
        #pragma unroll
        for (int j = 0; j < 4; ++j) WT[nq + j][k] = w4[j];
    }
    __syncthreads();

    const int wave = threadIdx.x >> 5;
    const int lane = threadIdx.x & 31;
    const int mt = blockIdx.x * 8 + wave;
    if (mt * 16 >= M) return;

    const int hi   = lane >> 4;             // half-wave select
    const int ksel = hi ? 8 : 0;            // A frag: lanes16-31 hold K+8..15 / K+24..31
    const int koff = hi ? 8 : 0;            // B frag: lanes16-31 hold K+8..15 / K+24..31
    const int row  = mt * 16 + (lane & 15);
    const u16* arow = A + (size_t)row * K;

    union { u32x4 u[2]; v16bf v; } af, bfr;
    v8f acc[4] = {};
    for (int k0 = 0; k0 < K; k0 += 32) {
        __builtin_prefetch(arow + k0 + 32, 0, 1);
        af.u[0] = *(const u32x4*)(arow + k0 + ksel);        // K = k0+ksel .. +7
        af.u[1] = *(const u32x4*)(arow + k0 + ksel + 16);   // K = k0+ksel+16 .. +23
        #pragma unroll
        for (int s = 0; s < 4; ++s) {                       // 4 column sub-tiles
            const u16* bp = &WT[(s << 4) + (lane & 15)][k0 + koff];
            bfr.u[0] = *(const u32x4*)(bp);
            bfr.u[1] = *(const u32x4*)(bp + 16);
            acc[s] = __builtin_amdgcn_wmma_f32_16x16x32_bf16(false, af.v, false, bfr.v,
                                                             (short)0, acc[s], false, false);
        }
    }

    #pragma unroll
    for (int s = 0; s < 4; ++s) {
        const int col = n0 + (s << 4) + (lane & 15);
        float b = 0.f;
        if (bias0) b += bias0[col];
        if (bias1) b += bias1[col];
        #pragma unroll
        for (int i = 0; i < 8; ++i) {
            int r = mt * 16 + i + (hi << 3);    // C/D layout: VGPR i -> M=i (+8 hi half)
            size_t o = (size_t)r * Nout + col;
            float v = acc[s][i] + b;
            if (addC)   v += C[o];
            if (doRelu) v = fmaxf(v, 0.f);
            if (C)  C[o]  = v;
            if (Cb) Cb[o] = f2bf(v);
        }
    }
}

// per (node, head): a_src / a_dst dot products over C=64, one wave each
__global__ __launch_bounds__(256)
void att_scores_kernel(const float* __restrict__ T, const float* __restrict__ aS,
                       const float* __restrict__ aD, float* __restrict__ outS,
                       float* __restrict__ outD, int Nn)
{
    int lane = threadIdx.x & 31;
    int id = blockIdx.x * 8 + (threadIdx.x >> 5);
    if (id >= Nn * HH) return;
    int h = id & 3;
    const float* t = T + (size_t)(id >> 2) * DD + h * CC;
    float t0 = t[lane], t1 = t[lane + 32];
    float ps = t0 * aS[h * CC + lane] + t1 * aS[h * CC + lane + 32];
    float pd = t0 * aD[h * CC + lane] + t1 * aD[h * CC + lane + 32];
    for (int o = 16; o > 0; o >>= 1) {
        ps += __shfl_xor(ps, o, 32);
        pd += __shfl_xor(pd, o, 32);
    }
    if (lane == 0) { outS[id] = ps; outD[id] = pd; }
}

__global__ void edge_max_kernel(const int* __restrict__ ei, const float* __restrict__ aS,
                                const float* __restrict__ aD, unsigned* __restrict__ emaxk,
                                float* __restrict__ albuf, int E, int Nn)
{
    int e = blockIdx.x * blockDim.x + threadIdx.x;
    if (e >= E + Nn) return;
    int s, d;
    if (e < E) { s = ei[e]; d = ei[E + e]; } else { s = d = e - E; }   // self loops appended
    #pragma unroll
    for (int h = 0; h < HH; ++h) {
        float v = aS[(size_t)s * HH + h] + aD[(size_t)d * HH + h];
        v = v > 0.f ? v : 0.2f * v;                                    // leaky relu
        albuf[(size_t)e * HH + h] = v;
        atomicMax(&emaxk[(size_t)d * HH + h], fkey(v));
    }
}

__global__ void edge_exp_kernel(const int* __restrict__ ei, const unsigned* __restrict__ emaxk,
                                float* __restrict__ albuf, float* __restrict__ denom,
                                int E, int Nn)
{
    int e = blockIdx.x * blockDim.x + threadIdx.x;
    if (e >= E + Nn) return;
    int d = (e < E) ? ei[E + e] : (e - E);
    #pragma unroll
    for (int h = 0; h < HH; ++h) {
        float m = fkey_inv(emaxk[(size_t)d * HH + h]);
        float a = __expf(albuf[(size_t)e * HH + h] - m);
        albuf[(size_t)e * HH + h] = a;
        atomicAdd(&denom[(size_t)d * HH + h], a);
    }
}

// one wave per edge: gather 256 floats from src row, weighted atomic scatter to dst
__global__ __launch_bounds__(256)
void edge_msg_kernel(const int* __restrict__ ei, const float* __restrict__ T,
                     const float* __restrict__ albuf, const float* __restrict__ denom,
                     float* __restrict__ G, int E, int Nn)
{
    int lane = threadIdx.x & 31;
    int e = blockIdx.x * 8 + (threadIdx.x >> 5);
    if (e >= E + Nn) return;
    int s, d;
    if (e < E) { s = ei[e]; d = ei[E + e]; } else { s = d = e - E; }
    int h = lane >> 3;                                   // 8 lanes per head (8*8=64 ch)
    float w = albuf[(size_t)e * HH + h] / denom[(size_t)d * HH + h];
    const float* sp = T + (size_t)s * DD + lane * 8;
    float* dp = G + (size_t)d * DD + lane * 8;
    #pragma unroll
    for (int j = 0; j < 8; ++j) atomicAdd(&dp[j], sp[j] * w);
}

// x_next = (1-sigmoid(beta)) * LN(h * x_new) + sigmoid(beta) * x_new ;
// xb (bf16 next-layer input) and xloc += x_next. One wave per node row.
__global__ __launch_bounds__(256)
void combine_kernel(const float* __restrict__ Hb, const float* __restrict__ X,
                    u16* __restrict__ xb, float* __restrict__ xloc,
                    const float* __restrict__ g, const float* __restrict__ bshift,
                    const float* __restrict__ bet, int Nn)
{
    int lane = threadIdx.x & 31;
    int r = blockIdx.x * 8 + (threadIdx.x >> 5);
    if (r >= Nn) return;
    const float* hr = Hb + (size_t)r * DD;
    const float* xr = X + (size_t)r * DD;
    float y[8], xv[8], s = 0.f, s2 = 0.f;
    #pragma unroll
    for (int j = 0; j < 8; ++j) {
        int c = lane + j * 32;
        xv[j] = xr[c];
        y[j] = hr[c] * xv[j];
        s += y[j]; s2 += y[j] * y[j];
    }
    for (int o = 16; o > 0; o >>= 1) { s += __shfl_xor(s, o, 32); s2 += __shfl_xor(s2, o, 32); }
    float mu  = s * (1.f / DD);
    float var = s2 * (1.f / DD) - mu * mu;
    float rs  = rsqrtf(var + 1e-5f);
    #pragma unroll
    for (int j = 0; j < 8; ++j) {
        int c = lane + j * 32;
        float ln   = (y[j] - mu) * rs * g[c] + bshift[c];
        float beta = 1.f / (1.f + __expf(-bet[c]));
        float xn   = (1.f - beta) * ln + beta * xv[j];
        xb[(size_t)r * DD + c]    = f2bf(xn);
        xloc[(size_t)r * DD + c] += xn;
    }
}

extern "C" void kernel_launch(void* const* d_in, const int* in_sizes, int n_in,
                              void* d_out, int out_size, void* d_ws, size_t ws_size,
                              hipStream_t stream) {
    (void)n_in; (void)out_size; (void)ws_size;
    const int N = in_sizes[0] / DD;
    const int E = in_sizes[1] / 2;

    const float* x_in    = (const float*)d_in[0];
    const int*   ei      = (const int*)  d_in[1];
    const float* Win     = (const float*)d_in[2];
    const float* b_in    = (const float*)d_in[3];
    const float* Wh      = (const float*)d_in[4];
    const float* bh      = (const float*)d_in[5];
    const float* Wg      = (const float*)d_in[6];
    const float* att_src = (const float*)d_in[7];
    const float* att_dst = (const float*)d_in[8];
    const float* bg      = (const float*)d_in[9];
    const float* Wl      = (const float*)d_in[10];
    const float* bl      = (const float*)d_in[11];
    const float* ln_g    = (const float*)d_in[12];
    const float* ln_b    = (const float*)d_in[13];
    const float* betas   = (const float*)d_in[14];
    const float* Wp      = (const float*)d_in[15];
    const float* bp      = (const float*)d_in[16];

    char* ws = (char*)d_ws;
    size_t off = 0;
    auto alloc = [&](size_t bytes) -> void* {
        void* p = (void*)(ws + off);
        off += (bytes + 255) & ~(size_t)255;
        return p;
    };
    float* Hbuf  = (float*)alloc((size_t)N * DD * 4);   // relu(x@Wh)
    float* Tbuf  = (float*)alloc((size_t)N * DD * 4);   // x@Wg (gather source)
    float* Gbuf  = (float*)alloc((size_t)N * DD * 4);   // GAT sum -> x_new
    float* xloc  = (float*)alloc((size_t)N * DD * 4);   // sum of per-layer x
    u16*   xb    = (u16*)  alloc((size_t)N * DD * 2);   // current x, bf16
    u16*   xinb  = (u16*)  alloc((size_t)N * DD * 2);   // bf16(x_in); reused for bf16(xloc)
    float* albuf = (float*)alloc((size_t)(E + N) * HH * 4);
    float* asrcB = (float*)alloc((size_t)N * HH * 4);
    float* adstB = (float*)alloc((size_t)N * HH * 4);
    float* denom = (float*)alloc((size_t)N * HH * 4);
    unsigned* emaxk = (unsigned*)alloc((size_t)N * HH * 4);
    u16* Winb = (u16*)alloc((size_t)DD * DD * 2);
    u16* Whb  = (u16*)alloc((size_t)3 * DD * DD * 2);
    u16* Wgb  = (u16*)alloc((size_t)3 * DD * DD * 2);
    u16* Wlb  = (u16*)alloc((size_t)3 * DD * DD * 2);
    u16* Wpb  = (u16*)alloc((size_t)DD * 64 * 2);

    const int TPB = 256;
    auto cdiv = [](long long a, long long b) { return (unsigned)((a + b - 1) / b); };

    // weights & input to bf16
    cvt_bf16_kernel<<<cdiv((long long)DD * DD, TPB), TPB, 0, stream>>>(Win, Winb, (size_t)DD * DD);
    cvt_bf16_kernel<<<cdiv(3LL * DD * DD, TPB), TPB, 0, stream>>>(Wh, Whb, (size_t)3 * DD * DD);
    cvt_bf16_kernel<<<cdiv(3LL * DD * DD, TPB), TPB, 0, stream>>>(Wg, Wgb, (size_t)3 * DD * DD);
    cvt_bf16_kernel<<<cdiv(3LL * DD * DD, TPB), TPB, 0, stream>>>(Wl, Wlb, (size_t)3 * DD * DD);
    cvt_bf16_kernel<<<cdiv((long long)DD * 64, TPB), TPB, 0, stream>>>(Wp, Wpb, (size_t)DD * 64);
    cvt_bf16_kernel<<<cdiv((long long)N * DD, TPB), TPB, 0, stream>>>(x_in, xinb, (size_t)N * DD);
    fill_f32_kernel<<<cdiv((long long)N * DD, TPB), TPB, 0, stream>>>(xloc, 0.f, (size_t)N * DD);

    const int mtiles = (N + 15) / 16;
    dim3 gemmGrid(cdiv(mtiles, 8), DD / 64);

    // x = x_in @ Win + b_in  (bf16 result only)
    gemm_bf16_wmma<<<gemmGrid, 256, 0, stream>>>(xinb, Winb, nullptr, xb,
                                                 b_in, nullptr, N, DD, DD, 0, 0);

    for (int i = 0; i < 3; ++i) {
        const u16* Whi = Whb + (size_t)i * DD * DD;
        const u16* Wgi = Wgb + (size_t)i * DD * DD;
        const u16* Wli = Wlb + (size_t)i * DD * DD;

        // h = relu(x @ Wh + bh)
        gemm_bf16_wmma<<<gemmGrid, 256, 0, stream>>>(xb, Whi, Hbuf, nullptr,
                                                     bh + i * DD, nullptr, N, DD, DD, 0, 1);
        // T = x @ Wg
        gemm_bf16_wmma<<<gemmGrid, 256, 0, stream>>>(xb, Wgi, Tbuf, nullptr,
                                                     nullptr, nullptr, N, DD, DD, 0, 0);
        // attention scores per (node, head)
        att_scores_kernel<<<cdiv((long long)N * HH, 8), 256, 0, stream>>>(
            Tbuf, att_src + (size_t)i * HH * CC, att_dst + (size_t)i * HH * CC,
            asrcB, adstB, N);
        // segment softmax + message passing
        fill_u32_kernel<<<cdiv((long long)N * HH, TPB), TPB, 0, stream>>>(emaxk, 0u, (size_t)N * HH);
        fill_f32_kernel<<<cdiv((long long)N * HH, TPB), TPB, 0, stream>>>(denom, 0.f, (size_t)N * HH);
        fill_f32_kernel<<<cdiv((long long)N * DD, TPB), TPB, 0, stream>>>(Gbuf, 0.f, (size_t)N * DD);
        edge_max_kernel<<<cdiv((long long)(E + N), TPB), TPB, 0, stream>>>(ei, asrcB, adstB, emaxk, albuf, E, N);
        edge_exp_kernel<<<cdiv((long long)(E + N), TPB), TPB, 0, stream>>>(ei, emaxk, albuf, denom, E, N);
        edge_msg_kernel<<<cdiv((long long)(E + N), 8), 256, 0, stream>>>(ei, Tbuf, albuf, denom, Gbuf, E, N);
        // x_new = relu(GATsum + bg + x @ Wl + bl)  (accumulate into Gbuf)
        gemm_bf16_wmma<<<gemmGrid, 256, 0, stream>>>(xb, Wli, Gbuf, nullptr,
                                                     bl + i * DD, bg + i * DD, N, DD, DD, 1, 1);
        // gated layernorm combine; writes bf16 x for next layer, xloc += x
        combine_kernel<<<cdiv(N, 8), 256, 0, stream>>>(Hbuf, Gbuf, xb, xloc,
                                                       ln_g + i * DD, ln_b + i * DD,
                                                       betas + i * DD, N);
    }

    // out = xloc @ Wp + bp
    cvt_bf16_kernel<<<cdiv((long long)N * DD, TPB), TPB, 0, stream>>>(xloc, xinb, (size_t)N * DD);
    dim3 pGrid(cdiv(mtiles, 8), 1);
    gemm_bf16_wmma<<<pGrid, 256, 0, stream>>>(xinb, Wpb, (float*)d_out, nullptr,
                                              bp, nullptr, N, DD, 64, 0, 0);
}
